// DrugGAT_5377299055264
// MI455X (gfx1250) — matrix operand order repaired and checked by reading, not compiled, not run on hardware
//
#include <hip/hip_runtime.h>
#include <hip/hip_bf16.h>

// ---------------------------------------------------------------------------
// Types for CDNA5 WMMA (wave32): v_wmma_f32_16x16x32_bf16
// ---------------------------------------------------------------------------
typedef __bf16 bf16_t;
typedef __attribute__((ext_vector_type(8)))  __bf16 v8bf;
typedef __attribute__((ext_vector_type(16))) __bf16 v16bf;
typedef __attribute__((ext_vector_type(8)))  float  v8f;
typedef __attribute__((ext_vector_type(4)))  unsigned int u32x4;
typedef __attribute__((ext_vector_type(4)))  int          i32x4;
typedef __attribute__((ext_vector_type(8)))  int          i32x8;

#define HAVE_TDM (__has_builtin(__builtin_amdgcn_tensor_load_to_lds) && \
                  __has_builtin(__builtin_amdgcn_s_wait_tensorcnt))

// ---------------------------------------------------------------------------
// float atomic max via signed/unsigned int ordering trick
// ---------------------------------------------------------------------------
__device__ __forceinline__ void atomicMaxF(float* addr, float v) {
  if (v >= 0.f) atomicMax((int*)addr, __float_as_int(v));
  else          atomicMin((unsigned int*)addr, (unsigned int)__float_as_int(v));
}

// ---------------------------------------------------------------------------
// f32 -> bf16 conversion
// ---------------------------------------------------------------------------
__global__ void cvt_f32_bf16(const float* __restrict__ in, bf16_t* __restrict__ out, long n) {
  long i = (long)blockIdx.x * blockDim.x + threadIdx.x;
  if (i < n) out[i] = (bf16_t)in[i];
}

// ---------------------------------------------------------------------------
// Dual GEMM: outl = A @ Wl + bl, outr = A @ Wr + br  (A: N x 128 bf16,
// W: 128 x 128 bf16, out: f32). Block = 256 threads = 8 waves; block owns a
// 16-row strip; wave w owns column tile n0 = 16*w for BOTH outputs, so each
// A fragment read from LDS feeds two WMMA chains (8 v_wmma / wave).
//
// A-tile (16x128 bf16 = 4KB) is staged in LDS by the Tensor Data Mover:
// wave 0 issues tensor_load_to_lds with a 2D D# (data_size=2B, tile 128x16,
// tensor_dim1 = Nrows so OOB tail rows are zero-filled by hardware), waits
// s_wait_tensorcnt 0, then the block barriers.
//
// Fragment layouts per CDNA5 ISA 7.12.2 (16-bit, wave32):
//  A 16x32 : lane&15 = row M; elems[0..7]=K(+0|+8 by lane half), [8..15]=K+16(+0|+8)
//  B 32x16 : lane&15 = col N; elems[0..7]=K(+0|+8 by lane half), [8..15]=K+16(+0|+8)
//  C/D f32 : VGPR i -> M = i (lanes 0-15) / 8+i (lanes 16-31); N = lane&15
// ---------------------------------------------------------------------------
__global__ __launch_bounds__(256) void gemm2_bf16_wmma(
    const bf16_t* __restrict__ A,
    const bf16_t* __restrict__ Wl, const bf16_t* __restrict__ Wr,
    const float* __restrict__ bl,  const float* __restrict__ br,
    float* __restrict__ outl, float* __restrict__ outr, int Nrows) {
  __shared__ bf16_t As[16 * 128];

  const int tid = threadIdx.x;
  const int m0  = blockIdx.x * 16;

#if HAVE_TDM
  if (tid < 32) {
    // ---- Tensor DMA descriptor (D#) -------------------------------------
    unsigned long long ga =
        (unsigned long long)(uintptr_t)(A + (size_t)m0 * 128);
    unsigned lds = (unsigned)(uintptr_t)(&As[0]);     // low 32 bits = LDS offset
    // Group 0: count=1 | lds_addr | global_addr[56:0] | type=2 (bits 127:126)
    u32x4 g0 = { 1u, lds, (unsigned)(ga & 0xFFFFFFFFu),
                 (unsigned)((ga >> 32) & 0x01FFFFFFu) | 0x80000000u };
    // Group 1: data_size=1 (2B); tensor_dim0=128; tensor_dim1=Nrows;
    //          tile_dim0=128; tile_dim1=16; tensor_dim0_stride=128
    i32x8 g1 = { (int)0x00010000u,                  // data_size=2B, mask=0
                 (int)(128u << 16),                 // tensor_dim0 lo16 @ [31:16]
                 (int)(((unsigned)Nrows & 0xFFFFu) << 16),  // dim0 hi16 | dim1 lo16
                 (int)((((unsigned)Nrows >> 16) & 0xFFFFu) | (128u << 16)), // dim1 hi | tile_dim0
                 16,                                // tile_dim1=16, tile_dim2=0
                 128,                               // tensor_dim0_stride lo32
                 0, 0 };
    i32x4 z4 = { 0, 0, 0, 0 };
#if __clang_major__ >= 23
    i32x8 z8 = { 0, 0, 0, 0, 0, 0, 0, 0 };
    __builtin_amdgcn_tensor_load_to_lds(g0, g1, z4, z4, z8, 0);
#else
    __builtin_amdgcn_tensor_load_to_lds(g0, g1, z4, z4, 0);
#endif
    __builtin_amdgcn_s_wait_tensorcnt(0);
  }
#else
  { // cooperative fallback: 2048 bf16, 8 per thread (16B chunks)
    int idx = tid * 8;
    int row = idx >> 7, col = idx & 127;
    int gr = m0 + row; if (gr >= Nrows) gr = Nrows - 1;
    *(v8bf*)&As[idx] = *(const v8bf*)&A[(size_t)gr * 128 + col];
  }
#endif
  __syncthreads();

  const int wave = tid >> 5;
  const int lane = tid & 31;
  const int half = lane >> 4;             // 0 | 1
  const int l15  = lane & 15;
  const int n0   = wave * 16;

  v8f accL = {}, accR = {};
#pragma unroll
  for (int k0 = 0; k0 < 128; k0 += 32) {
    // A fragment from LDS (shared by both chains)
    v8bf a_lo = *(const v8bf*)&As[l15 * 128 + k0 + 8 * half];
    v8bf a_hi = *(const v8bf*)&As[l15 * 128 + k0 + 16 + 8 * half];
    v16bf a = __builtin_shufflevector(a_lo, a_hi,
                0, 1, 2, 3, 4, 5, 6, 7, 8, 9, 10, 11, 12, 13, 14, 15);
    // B fragments: col n = n0 + l15; elems[0..7]=K(+0|+8 by lane half), [8..15]=K+16(+0|+8)
    const bf16_t* blp = Wl + (size_t)(k0 + 8 * half) * 128 + n0 + l15;
    const bf16_t* brp = Wr + (size_t)(k0 + 8 * half) * 128 + n0 + l15;
    v16bf bL, bR;
#pragma unroll
    for (int s = 0; s < 8; ++s) {
      bL[s] = blp[(size_t)s * 128]; bL[8 + s] = blp[(size_t)(16 + s) * 128];
      bR[s] = brp[(size_t)s * 128]; bR[8 + s] = brp[(size_t)(16 + s) * 128];
    }

    accL = __builtin_amdgcn_wmma_f32_16x16x32_bf16(false, a, false, bL,
                                                   (short)0, accL, false, false);
    accR = __builtin_amdgcn_wmma_f32_16x16x32_bf16(false, a, false, bR,
                                                   (short)0, accR, false, false);
  }

  const float bvl = bl[n0 + l15];
  const float bvr = br[n0 + l15];
  float* pl = outl + (size_t)(m0 + 8 * half) * 128 + n0 + l15;
  float* pr = outr + (size_t)(m0 + 8 * half) * 128 + n0 + l15;
  if (m0 + 16 <= Nrows) {           // fast path: full tile in range
#pragma unroll
    for (int i = 0; i < 8; ++i) {
      pl[(size_t)i * 128] = accL[i] + bvl;
      pr[(size_t)i * 128] = accR[i] + bvr;
    }
  } else {
#pragma unroll
    for (int i = 0; i < 8; ++i) {
      int r = m0 + i + 8 * half;
      if (r < Nrows) {
        outl[(size_t)r * 128 + n0 + l15] = accL[i] + bvl;
        outr[(size_t)r * 128 + n0 + l15] = accR[i] + bvr;
      }
    }
  }
}

// ---------------------------------------------------------------------------
// Node-state init: m = -inf-ish, s = 0   (N*H elements)
// ---------------------------------------------------------------------------
__global__ void init_ms(float* __restrict__ mbuf, float* __restrict__ sbuf, int n) {
  int i = blockIdx.x * blockDim.x + threadIdx.x;
  if (i < n) { mbuf[i] = -3.0e38f; sbuf[i] = 0.f; }
}

// Aggregate buffer init with bias (free bias-add): out[r,c] = bias[c]
__global__ void init_out_bias(float* __restrict__ out, const float* __restrict__ bias, long n) {
  long i = (long)blockIdx.x * blockDim.x + threadIdx.x;
  if (i < n) out[i] = bias[i & 127];
}

// ---------------------------------------------------------------------------
// Pass 1: per (edge, head) logits + segment atomic-max over dst
// ---------------------------------------------------------------------------
__global__ void edge_logits_max(const float* __restrict__ xl, const float* __restrict__ xr,
                                const float* __restrict__ att, const int* __restrict__ ei,
                                int E, int Etot, int hlog, int C,
                                float* __restrict__ logits, float* __restrict__ mbuf) {
  int gid = blockIdx.x * blockDim.x + threadIdx.x;
  if (gid >= (Etot << hlog)) return;
  int e = gid >> hlog, h = gid & ((1 << hlog) - 1);
  int src, dst;
  if (e < E) { src = ei[e]; dst = ei[E + e]; } else { src = dst = e - E; }

  const float4* pl = (const float4*)(xl + (size_t)src * 128 + h * C);
  const float4* pr = (const float4*)(xr + (size_t)dst * 128 + h * C);
  const float4* pa = (const float4*)(att + h * C);
  float acc = 0.f;
  const int C4 = C >> 2;
#pragma unroll 8
  for (int c = 0; c < C4; ++c) {
    float4 a = pl[c], b = pr[c], w = pa[c];
    float v0 = a.x + b.x; v0 = v0 > 0.f ? v0 : 0.2f * v0;
    float v1 = a.y + b.y; v1 = v1 > 0.f ? v1 : 0.2f * v1;
    float v2 = a.z + b.z; v2 = v2 > 0.f ? v2 : 0.2f * v2;
    float v3 = a.w + b.w; v3 = v3 > 0.f ? v3 : 0.2f * v3;
    acc += w.x * v0 + w.y * v1 + w.z * v2 + w.w * v3;
  }
  logits[gid] = acc;
  atomicMaxF(mbuf + ((size_t)dst << hlog) + h, acc);
}

// ---------------------------------------------------------------------------
// Pass 2: a = exp(logit - m[dst]); s[dst] += a   (in place over logits)
// ---------------------------------------------------------------------------
__global__ void edge_exp_sum(const int* __restrict__ ei, int E, int Etot, int hlog,
                             const float* __restrict__ mbuf, float* __restrict__ logits,
                             float* __restrict__ sbuf) {
  int gid = blockIdx.x * blockDim.x + threadIdx.x;
  if (gid >= (Etot << hlog)) return;
  int e = gid >> hlog, h = gid & ((1 << hlog) - 1);
  int dst = (e < E) ? ei[E + e] : (e - E);
  float a = __expf(logits[gid] - mbuf[((size_t)dst << hlog) + h]);
  logits[gid] = a;
  atomicAdd(sbuf + ((size_t)dst << hlog) + h, a);
}

// ---------------------------------------------------------------------------
// Pass 2b: alpha = a / (s[dst] + eps)   (in place over a)
// ---------------------------------------------------------------------------
__global__ void edge_alpha(const int* __restrict__ ei, int E, int Etot, int hlog,
                           const float* __restrict__ sbuf, float* __restrict__ abuf) {
  int gid = blockIdx.x * blockDim.x + threadIdx.x;
  if (gid >= (Etot << hlog)) return;
  int e = gid >> hlog, h = gid & ((1 << hlog) - 1);
  int dst = (e < E) ? ei[E + e] : (e - E);
  abuf[gid] = abuf[gid] / (sbuf[((size_t)dst << hlog) + h] + 1e-16f);
}

// ---------------------------------------------------------------------------
// Pass 3: out[dst, c] += alpha[e, c/C] * xl[src, c]   (32 threads/edge)
// ---------------------------------------------------------------------------
__global__ void edge_aggregate(const int* __restrict__ ei, int E, int Etot, int hlog, int C,
                               const float* __restrict__ xl, const float* __restrict__ abuf,
                               float* __restrict__ out) {
  int gid = blockIdx.x * blockDim.x + threadIdx.x;
  if (gid >= Etot * 32) return;
  int e = gid >> 5, t = gid & 31;
  int src, dst;
  if (e < E) { src = ei[e]; dst = ei[E + e]; } else { src = dst = e - E; }
  const float* srow = xl + (size_t)src * 128;
  __builtin_prefetch(srow + t, 0, 0);     // global_prefetch_b8 on gathered row
#pragma unroll
  for (int j = 0; j < 4; ++j) {
    int c = t + 32 * j;
    int h = c / C;
    float alpha = abuf[((size_t)e << hlog) + h];
    atomicAdd(out + (size_t)dst * 128 + c, alpha * srow[c]);
  }
}

// ---------------------------------------------------------------------------
// Fused BatchNorm1d(eval) + ELU, in place
// ---------------------------------------------------------------------------
__global__ void bn_elu(float* __restrict__ x, const float* __restrict__ g,
                       const float* __restrict__ be, const float* __restrict__ mean,
                       const float* __restrict__ var, long n) {
  long i = (long)blockIdx.x * blockDim.x + threadIdx.x;
  if (i >= n) return;
  int c = (int)(i & 127);
  float y = (x[i] - mean[c]) * rsqrtf(var[c] + 1e-5f) * g[c] + be[c];
  x[i] = y > 0.f ? y : (__expf(y) - 1.f);
}

// ---------------------------------------------------------------------------
// Host-side: one GATv2 layer
// ---------------------------------------------------------------------------
static void run_gatv2_layer(const float* cur, const float* Wl, const float* bl,
                            const float* Wr, const float* br, const float* att,
                            const float* bias, float* agg,
                            bf16_t* xb, bf16_t* wlb, bf16_t* wrb,
                            float* xl, float* xr, float* logit,
                            float* mbuf, float* sbuf,
                            const int* ei, int N, int E, int Etot, int hlog, int C,
                            hipStream_t stream) {
  const long NF = (long)N * 128;
  const int  blk = 256;
  const int  H = 1 << hlog;

  cvt_f32_bf16<<<(int)((NF + blk - 1) / blk), blk, 0, stream>>>(cur, xb, NF);
  cvt_f32_bf16<<<(16384 + blk - 1) / blk, blk, 0, stream>>>(Wl, wlb, 16384);
  cvt_f32_bf16<<<(16384 + blk - 1) / blk, blk, 0, stream>>>(Wr, wrb, 16384);

  gemm2_bf16_wmma<<<(N + 15) / 16, 256, 0, stream>>>(xb, wlb, wrb, bl, br, xl, xr, N);

  init_ms<<<(N * H + blk - 1) / blk, blk, 0, stream>>>(mbuf, sbuf, N * H);
  init_out_bias<<<(int)((NF + blk - 1) / blk), blk, 0, stream>>>(agg, bias, NF);

  const int totEH = Etot * H;
  edge_logits_max<<<(totEH + blk - 1) / blk, blk, 0, stream>>>(
      xl, xr, att, ei, E, Etot, hlog, C, logit, mbuf);
  edge_exp_sum<<<(totEH + blk - 1) / blk, blk, 0, stream>>>(
      ei, E, Etot, hlog, mbuf, logit, sbuf);
  edge_alpha<<<(totEH + blk - 1) / blk, blk, 0, stream>>>(
      ei, E, Etot, hlog, sbuf, logit);
  edge_aggregate<<<(Etot * 32 + blk - 1) / blk, blk, 0, stream>>>(
      ei, E, Etot, hlog, C, xl, logit, agg);
}

extern "C" void kernel_launch(void* const* d_in, const int* in_sizes, int n_in,
                              void* d_out, int out_size, void* d_ws, size_t ws_size,
                              hipStream_t stream) {
  const float* x    = (const float*)d_in[0];
  const float* W1l  = (const float*)d_in[1];  const float* b1l  = (const float*)d_in[2];
  const float* W1r  = (const float*)d_in[3];  const float* b1r  = (const float*)d_in[4];
  const float* att1 = (const float*)d_in[5];  const float* bias1= (const float*)d_in[6];
  const float* g1   = (const float*)d_in[7];  const float* be1  = (const float*)d_in[8];
  const float* m1   = (const float*)d_in[9];  const float* v1   = (const float*)d_in[10];
  const float* W2l  = (const float*)d_in[11]; const float* b2l  = (const float*)d_in[12];
  const float* W2r  = (const float*)d_in[13]; const float* b2r  = (const float*)d_in[14];
  const float* att2 = (const float*)d_in[15]; const float* bias2= (const float*)d_in[16];
  const float* g2   = (const float*)d_in[17]; const float* be2  = (const float*)d_in[18];
  const float* m2   = (const float*)d_in[19]; const float* v2   = (const float*)d_in[20];
  const float* W3l  = (const float*)d_in[21]; const float* b3l  = (const float*)d_in[22];
  const float* W3r  = (const float*)d_in[23]; const float* b3r  = (const float*)d_in[24];
  const float* att3 = (const float*)d_in[25]; const float* bias3= (const float*)d_in[26];
  const int*   ei   = (const int*)d_in[27];

  const int N    = in_sizes[0] / 128;
  const int E    = in_sizes[27] / 2;
  const int Etot = E + N;

  // Workspace carve (256B aligned)
  char* ws = (char*)d_ws;
  size_t off = 0;
  auto carve = [&](size_t bytes) -> char* {
    char* p = ws + off;
    off += (bytes + 255) & ~(size_t)255;
    return p;
  };
  bf16_t* xb    = (bf16_t*)carve((size_t)N * 128 * sizeof(bf16_t));
  bf16_t* wlb   = (bf16_t*)carve(128 * 128 * sizeof(bf16_t));
  bf16_t* wrb   = (bf16_t*)carve(128 * 128 * sizeof(bf16_t));
  float*  xl    = (float*)carve((size_t)N * 128 * 4);
  float*  xr    = (float*)carve((size_t)N * 128 * 4);
  float*  logit = (float*)carve((size_t)Etot * 4 * 4);
  float*  mbuf  = (float*)carve((size_t)N * 4 * 4);
  float*  sbuf  = (float*)carve((size_t)N * 4 * 4);
  float*  h1    = (float*)carve((size_t)N * 128 * 4);
  float*  h2    = (float*)carve((size_t)N * 128 * 4);

  const long NF = (long)N * 128;
  const int  blk = 256;

  // Layer 1: H=4 (hlog=2), C=32, concat
  run_gatv2_layer(x, W1l, b1l, W1r, b1r, att1, bias1, h1,
                  xb, wlb, wrb, xl, xr, logit, mbuf, sbuf,
                  ei, N, E, Etot, 2, 32, stream);
  bn_elu<<<(int)((NF + blk - 1) / blk), blk, 0, stream>>>(h1, g1, be1, m1, v1, NF);

  // Layer 2: H=4, C=32, concat
  run_gatv2_layer(h1, W2l, b2l, W2r, b2r, att2, bias2, h2,
                  xb, wlb, wrb, xl, xr, logit, mbuf, sbuf,
                  ei, N, E, Etot, 2, 32, stream);
  bn_elu<<<(int)((NF + blk - 1) / blk), blk, 0, stream>>>(h2, g2, be2, m2, v2, NF);

  // Layer 3: H=1 (hlog=0), C=128, mean over 1 head == identity; bias in init
  run_gatv2_layer(h2, W3l, b3l, W3r, b3r, att3, bias3, (float*)d_out,
                  xb, wlb, wrb, xl, xr, logit, mbuf, sbuf,
                  ei, N, E, Etot, 0, 128, stream);
}
